// SequenceDiff_22351009808573
// MI455X (gfx1250) — matrix operand order repaired and checked by reading, not compiled, not run on hardware
//
#include <hip/hip_runtime.h>
#include <hip/hip_bf16.h>
#include <math.h>

typedef _Float16 v8h  __attribute__((ext_vector_type(8)));
typedef _Float16 v16h __attribute__((ext_vector_type(16)));
typedef float    v8f  __attribute__((ext_vector_type(8)));

union HFrag { v16h v; v8h h[2]; };

#define L_SEQ   256
#define L2_SEQ  512
#define BATCH   16
#define HEADS   12
#define DDIM    768
#define HHID    384
#define GATES   1536   // 4*HHID
#define KDIM    1536   // 2*DDIM
#define NSEQ    192    // BATCH*HEADS
#define SLIDERN 6
#define MROWS   1152   // NSEQ*SLIDERN

// ---------------------------------------------------------------------------
// Kernel 1: masked row-sums of the two diagonal attention blocks + argmax.
// One block per (b,h). 8 waves x 32 rows each, float4 coalesced reads.
// ---------------------------------------------------------------------------
__global__ void __launch_bounds__(256)
k_master(const int* __restrict__ sent, const float* __restrict__ att,
         int* __restrict__ masterA, int* __restrict__ masterB)
{
    __shared__ unsigned char km[L2_SEQ];
    __shared__ float rsum[L_SEQ];
    __shared__ int   ridx[L_SEQ];

    int q = blockIdx.x;            // b*HEADS + h
    int b = q / HEADS;
    int tid = threadIdx.x;
    int lane = tid & 31, wave = tid >> 5;

    for (int j = tid; j < L2_SEQ; j += 256)
        km[j] = (sent[b * L2_SEQ + j] == 102) ? 1 : 0;
    __syncthreads();

    const float* base = att + (size_t)q * L2_SEQ * L2_SEQ;

    for (int half = 0; half < 2; ++half) {
        int off = half * L_SEQ;    // row & col offset of diagonal block
        for (int r = 0; r < 32; ++r) {
            int i = wave * 32 + r;
            const float* row = base + (size_t)(off + i) * L2_SEQ + off;
            const unsigned char* kmp = km + off;
            int j0 = lane * 4;
            float4 v0 = *(const float4*)(row + j0);
            float4 v1 = *(const float4*)(row + j0 + 128);
            float s = 0.f;
            s += kmp[j0 + 0]   ? 1e-9f : v0.x;
            s += kmp[j0 + 1]   ? 1e-9f : v0.y;
            s += kmp[j0 + 2]   ? 1e-9f : v0.z;
            s += kmp[j0 + 3]   ? 1e-9f : v0.w;
            s += kmp[j0 + 128] ? 1e-9f : v1.x;
            s += kmp[j0 + 129] ? 1e-9f : v1.y;
            s += kmp[j0 + 130] ? 1e-9f : v1.z;
            s += kmp[j0 + 131] ? 1e-9f : v1.w;
            for (int m = 16; m >= 1; m >>= 1) s += __shfl_xor(s, m, 32);
            if (lane == 0) rsum[i] = s;
        }
        __syncthreads();
        ridx[tid] = tid;
        __syncthreads();
        for (int st = 128; st >= 1; st >>= 1) {
            if (tid < st) {
                if (rsum[tid + st] > rsum[tid]) {
                    rsum[tid] = rsum[tid + st];
                    ridx[tid] = ridx[tid + st];
                }
            }
            __syncthreads();
        }
        if (tid == 0) (half ? masterB : masterA)[q] = ridx[0];
        __syncthreads();
    }
}

// ---------------------------------------------------------------------------
// Kernel 2: window starts, gather indices, touched/replace flags. 1 block / b.
// ---------------------------------------------------------------------------
__device__ __forceinline__ int win_start(int pos)
{
    int p = pos;
    if (p == 0)         p = 1;
    if (p == L_SEQ - 1) p = L_SEQ - 2;
    int l = 3;
    if (p - 3 <= 0) l = p - 1;
    int r = SLIDERN - l;
    if (p + r >= L_SEQ - 1) { int r2 = L_SEQ - p - 2; l = SLIDERN - r2; }
    return p - l;
}

__global__ void __launch_bounds__(512)
k_windows(const int* __restrict__ masterA, const int* __restrict__ masterB,
          const float* __restrict__ mask,
          int* __restrict__ idxA, int* __restrict__ idxB, int* __restrict__ replace)
{
    __shared__ int touched[L2_SEQ];
    int b = blockIdx.x, t = threadIdx.x;
    touched[t] = 0;
    __syncthreads();
    if (t < HEADS) {
        int q = b * HEADS + t;
        int sA = win_start(masterA[q]);
        int sB = win_start(masterB[q]);
        for (int s = 0; s < SLIDERN; ++s) {
            int pA = sA + s;
            int pB = L_SEQ + sB + s;
            idxA[q * SLIDERN + s] = pA;
            idxB[q * SLIDERN + s] = pB;
            touched[pA] = 1;
            touched[pB] = 1;
        }
    }
    __syncthreads();
    replace[b * L2_SEQ + t] = (touched[t] && mask[b * L2_SEQ + t] == 0.f) ? 1 : 0;
}

// ---------------------------------------------------------------------------
// Kernel: f32 -> f16 convert (weights).
// ---------------------------------------------------------------------------
__global__ void k_f32_to_f16(const float* __restrict__ src, _Float16* __restrict__ dst, int n)
{
    int i = blockIdx.x * 256 + threadIdx.x;
    if (i < n) dst[i] = (_Float16)src[i];
}

// ---------------------------------------------------------------------------
// Kernel 3a: gather slide windows into X f16, rows = q*6+s, cols = [A|B] dims.
// ---------------------------------------------------------------------------
__global__ void k_gather(const float* __restrict__ logits,
                         const int* __restrict__ idxA, const int* __restrict__ idxB,
                         const int* __restrict__ replace, _Float16* __restrict__ X)
{
    int i = blockIdx.x * 256 + threadIdx.x;
    if (i >= MROWS * KDIM) return;
    int row = i / KDIM, col = i % KDIM;
    int q = row / SLIDERN, s = row % SLIDERN;
    int b = q / HEADS;
    int pos, d;
    if (col < DDIM) { pos = idxA[q * SLIDERN + s]; d = col; }
    else            { pos = idxB[q * SLIDERN + s]; d = col - DDIM; }
    float v = replace[b * L2_SEQ + pos] ? 1e-9f
                                        : logits[((size_t)b * L2_SEQ + pos) * DDIM + d];
    X[i] = (_Float16)v;
}

// ---------------------------------------------------------------------------
// Kernel 3b: WMMA GEMM, G = X(1152x1536) * W^T + (b_ih+b_hh).
// W is f16 in the original [n][k] layout, so B fragments are k-contiguous.
// Block tile 128x128, 8 waves, each wave 32x64 (2x4 16x16 tiles), K step 32.
// ---------------------------------------------------------------------------
__global__ void __launch_bounds__(256)
k_gemm_gates(const _Float16* __restrict__ X, const _Float16* __restrict__ W,
             const float* __restrict__ b_ih, const float* __restrict__ b_hh,
             float* __restrict__ G)
{
    __shared__ _Float16 At[128][32];
    __shared__ _Float16 Bt[128][32];

    int tid  = threadIdx.x;
    int lane = tid & 31, wave = tid >> 5;
    int nBase = blockIdx.x * 128;
    int mBase = blockIdx.y * 128;
    int wm = (wave & 3) * 32;     // 2 m-tiles of 16
    int wn = (wave >> 2) * 64;    // 4 n-tiles of 16

    v8f acc[2][4];
    for (int i = 0; i < 2; ++i)
        for (int j = 0; j < 4; ++j)
            for (int r = 0; r < 8; ++r) acc[i][j][r] = 0.f;

    int srow = tid >> 1;          // 0..127
    int shk  = (tid & 1) * 16;    // 0 or 16 halves

    int mr  = lane & 15;
    int kbA = (lane >> 4) * 8;    // A frag: k in {kbA..kbA+7, kbA+16..kbA+23}
    int kbB = (lane >> 4) * 8;    // B frag: k in {kbB..kbB+7, kbB+16..kbB+23}

    for (int kc = 0; kc < KDIM; kc += 32) {
        __syncthreads();
        {
            const v8h* gA = (const v8h*)(X + (size_t)(mBase + srow) * KDIM + kc + shk);
            *(v8h*)&At[srow][shk]     = gA[0];
            *(v8h*)&At[srow][shk + 8] = gA[1];
            const v8h* gB = (const v8h*)(W + (size_t)(nBase + srow) * KDIM + kc + shk);
            *(v8h*)&Bt[srow][shk]     = gB[0];
            *(v8h*)&Bt[srow][shk + 8] = gB[1];
            if (kc + 32 < KDIM) {  // hint next A/B chunk into cache
                __builtin_prefetch(X + (size_t)(mBase + srow) * KDIM + kc + 32 + shk, 0, 1);
                __builtin_prefetch(W + (size_t)(nBase + srow) * KDIM + kc + 32 + shk, 0, 1);
            }
        }
        __syncthreads();

        HFrag a[2], bf[4];
        for (int i = 0; i < 2; ++i) {
            int m = wm + i * 16 + mr;
            a[i].h[0] = *(const v8h*)&At[m][kbA];
            a[i].h[1] = *(const v8h*)&At[m][kbA + 16];
        }
        for (int j = 0; j < 4; ++j) {
            int n = wn + j * 16 + mr;
            bf[j].h[0] = *(const v8h*)&Bt[n][kbB];
            bf[j].h[1] = *(const v8h*)&Bt[n][kbB + 16];
        }
        for (int i = 0; i < 2; ++i)
            for (int j = 0; j < 4; ++j)
                acc[i][j] = __builtin_amdgcn_wmma_f32_16x16x32_f16(
                    false, a[i].v, false, bf[j].v, (short)0, acc[i][j], false, false);
    }

    for (int i = 0; i < 2; ++i) {
        int m0 = mBase + wm + i * 16 + (lane >> 4) * 8;
        for (int j = 0; j < 4; ++j) {
            int n = nBase + wn + j * 16 + mr;
            float bias = b_ih[n] + b_hh[n];
            for (int r = 0; r < 8; ++r)
                G[(size_t)(m0 + r) * GATES + n] = acc[i][j][r] + bias;
        }
    }
}

// ---------------------------------------------------------------------------
// Kernel 4: recurrent LSTM. 24 blocks = 12 seq-tiles x 2 directions.
// Block holds h (16x384 f16) in LDS, c in registers (WMMA C layout), runs
// all 6 steps. Per step: G = Gpre + h @ Whh^T via WMMA, then gates.
// Wave w owns units [w*48, w*48+48): 4 gates x 3 unit-tiles of 16.
// ---------------------------------------------------------------------------
__global__ void __launch_bounds__(256)
k_lstm(const float* __restrict__ Gf, const float* __restrict__ Gr,
       const _Float16* __restrict__ WhhF, const _Float16* __restrict__ WhhR,
       float* __restrict__ hsF, float* __restrict__ hsR)
{
    __shared__ _Float16 hbuf[16][HHID];

    int blk = blockIdx.x;
    int dir = blk / 12;
    int seqBase = (blk % 12) * 16;
    const float*    Gp  = dir ? Gr   : Gf;
    const _Float16* Whh = dir ? WhhR : WhhF;
    float*          hs  = dir ? hsR  : hsF;

    int tid = threadIdx.x, lane = tid & 31, wave = tid >> 5;
    int mr = lane & 15, hi = lane >> 4;

    for (int i = tid; i < 16 * HHID; i += 256) ((_Float16*)hbuf)[i] = (_Float16)0.f;

    v8f c[3];
    for (int ut = 0; ut < 3; ++ut)
        for (int r = 0; r < 8; ++r) c[ut][r] = 0.f;
    __syncthreads();

    for (int st = 0; st < SLIDERN; ++st) {
        int s = dir ? (SLIDERN - 1 - st) : st;

        v8f acc[4][3];
        for (int g = 0; g < 4; ++g)
            for (int ut = 0; ut < 3; ++ut) {
                int n = g * HHID + wave * 48 + ut * 16 + mr;
                for (int r = 0; r < 8; ++r) {
                    int m = hi * 8 + r;
                    acc[g][ut][r] =
                        Gp[(size_t)((seqBase + m) * SLIDERN + s) * GATES + n];
                }
            }

        for (int kc = 0; kc < HHID; kc += 32) {
            HFrag a;
            a.h[0] = *(const v8h*)&hbuf[mr][kc + hi * 8];
            a.h[1] = *(const v8h*)&hbuf[mr][kc + hi * 8 + 16];
            for (int g = 0; g < 4; ++g)
                for (int ut = 0; ut < 3; ++ut) {
                    int n = g * HHID + wave * 48 + ut * 16 + mr;
                    HFrag bf;
                    const v8h* pb = (const v8h*)(Whh + (size_t)n * HHID + kc + hi * 8);
                    bf.h[0] = pb[0];
                    bf.h[1] = pb[2];
                    acc[g][ut] = __builtin_amdgcn_wmma_f32_16x16x32_f16(
                        false, a.v, false, bf.v, (short)0, acc[g][ut], false, false);
                }
        }

        __syncthreads();   // all reads of hbuf done before overwrite
        for (int ut = 0; ut < 3; ++ut) {
            int u = wave * 48 + ut * 16 + mr;
            for (int r = 0; r < 8; ++r) {
                int m = hi * 8 + r;
                float iv = 1.f / (1.f + __expf(-acc[0][ut][r]));
                float fv = 1.f / (1.f + __expf(-acc[1][ut][r]));
                float gv = tanhf(acc[2][ut][r]);
                float ov = 1.f / (1.f + __expf(-acc[3][ut][r]));
                float cn = fv * c[ut][r] + iv * gv;
                c[ut][r] = cn;
                float hv = ov * tanhf(cn);
                hbuf[m][u] = (_Float16)hv;
                hs[(size_t)((seqBase + m) * SLIDERN + s) * HHID + u] = hv;
            }
        }
        __syncthreads();   // h visible to all waves next step
    }
}

// ---------------------------------------------------------------------------
// Kernel 5: compose means + logits mean + FC(2) + softmax. 1 block per b.
// ---------------------------------------------------------------------------
__global__ void __launch_bounds__(256)
k_head(const float* __restrict__ hsF, const float* __restrict__ hsR,
       const float* __restrict__ logits, const int* __restrict__ replace,
       const float* __restrict__ fcw, const float* __restrict__ fcb,
       float* __restrict__ out)
{
    __shared__ float red0[256];
    __shared__ float red1[256];
    int b = blockIdx.x, t = threadIdx.x;
    float p0 = 0.f, p1 = 0.f;

    // rep part: 12*768 dims, rep = mean over 6 steps of concat(fwd, rev)
    for (int idx = t; idx < HEADS * DDIM; idx += 256) {
        int h12 = idx / DDIM, d = idx % DDIM;
        int q = b * HEADS + h12;
        const float* src = (d < HHID) ? (hsF + (size_t)q * SLIDERN * HHID + d)
                                      : (hsR + (size_t)q * SLIDERN * HHID + (d - HHID));
        float m = 0.f;
        for (int s = 0; s < SLIDERN; ++s) m += src[s * HHID];
        m *= (1.f / 6.f);
        p0 += m * fcw[idx];
        p1 += m * fcw[(HEADS + 1) * DDIM + idx];
    }
    // logits_rep part: mean over 512 positions (coalesced over d)
    for (int d = t; d < DDIM; d += 256) {
        float sum = 0.f;
        for (int p = 0; p < L2_SEQ; ++p) {
            float v = replace[b * L2_SEQ + p] ? 1e-9f
                                              : logits[((size_t)b * L2_SEQ + p) * DDIM + d];
            sum += v;
        }
        float lr = sum * (1.f / 512.f);
        p0 += lr * fcw[HEADS * DDIM + d];
        p1 += lr * fcw[(HEADS + 1) * DDIM + HEADS * DDIM + d];
    }

    red0[t] = p0; red1[t] = p1;
    __syncthreads();
    for (int st = 128; st >= 1; st >>= 1) {
        if (t < st) { red0[t] += red0[t + st]; red1[t] += red1[t + st]; }
        __syncthreads();
    }
    if (t == 0) {
        float z0 = red0[0] + fcb[0];
        float z1 = red1[0] + fcb[1];
        float mx = fmaxf(z0, z1);
        float e0 = __expf(z0 - mx), e1 = __expf(z1 - mx);
        float inv = 1.f / (e0 + e1);
        out[b * 2 + 0] = e0 * inv;
        out[b * 2 + 1] = e1 * inv;
    }
}

// ---------------------------------------------------------------------------
extern "C" void kernel_launch(void* const* d_in, const int* in_sizes, int n_in,
                              void* d_out, int out_size, void* d_ws, size_t ws_size,
                              hipStream_t stream)
{
    const int*   sent    = (const int*)d_in[0];
    const float* att     = (const float*)d_in[1];
    const float* logits  = (const float*)d_in[2];
    const float* mask    = (const float*)d_in[3];
    const float* w_ih_f  = (const float*)d_in[4];
    const float* w_hh_f  = (const float*)d_in[5];
    const float* b_ih_f  = (const float*)d_in[6];
    const float* b_hh_f  = (const float*)d_in[7];
    const float* w_ih_r  = (const float*)d_in[8];
    const float* w_hh_r  = (const float*)d_in[9];
    const float* b_ih_r  = (const float*)d_in[10];
    const float* b_hh_r  = (const float*)d_in[11];
    const float* fcw     = (const float*)d_in[12];
    const float* fcb     = (const float*)d_in[13];

    char* ws = (char*)d_ws;
    size_t off = 0;
    auto alloc = [&](size_t bytes) -> void* {
        off = (off + 255) & ~(size_t)255;
        void* p = ws + off;
        off += bytes;
        return p;
    };

    int*      masterA = (int*)alloc(NSEQ * 4);
    int*      masterB = (int*)alloc(NSEQ * 4);
    int*      idxA    = (int*)alloc(NSEQ * SLIDERN * 4);
    int*      idxB    = (int*)alloc(NSEQ * SLIDERN * 4);
    int*      replace = (int*)alloc(BATCH * L2_SEQ * 4);
    _Float16* wih16f  = (_Float16*)alloc((size_t)GATES * KDIM * 2);
    _Float16* wih16r  = (_Float16*)alloc((size_t)GATES * KDIM * 2);
    _Float16* whh16f  = (_Float16*)alloc((size_t)GATES * HHID * 2);
    _Float16* whh16r  = (_Float16*)alloc((size_t)GATES * HHID * 2);
    _Float16* X16     = (_Float16*)alloc((size_t)MROWS * KDIM * 2);
    float*    Gfwd    = (float*)alloc((size_t)MROWS * GATES * 4);
    float*    Grev    = (float*)alloc((size_t)MROWS * GATES * 4);
    float*    hsF     = (float*)alloc((size_t)NSEQ * SLIDERN * HHID * 4);
    float*    hsR     = (float*)alloc((size_t)NSEQ * SLIDERN * HHID * 4);

    const int nWih = GATES * KDIM;   // 2359296
    const int nWhh = GATES * HHID;   // 589824
    k_f32_to_f16<<<(nWih + 255) / 256, 256, 0, stream>>>(w_ih_f, wih16f, nWih);
    k_f32_to_f16<<<(nWih + 255) / 256, 256, 0, stream>>>(w_ih_r, wih16r, nWih);
    k_f32_to_f16<<<(nWhh + 255) / 256, 256, 0, stream>>>(w_hh_f, whh16f, nWhh);
    k_f32_to_f16<<<(nWhh + 255) / 256, 256, 0, stream>>>(w_hh_r, whh16r, nWhh);

    k_master<<<NSEQ, 256, 0, stream>>>(sent, att, masterA, masterB);
    k_windows<<<BATCH, 512, 0, stream>>>(masterA, masterB, mask, idxA, idxB, replace);

    const int nX = MROWS * KDIM;
    k_gather<<<(nX + 255) / 256, 256, 0, stream>>>(logits, idxA, idxB, replace, X16);

    dim3 ggrid(KDIM / 128, MROWS / 128);   // 12 x 9
    k_gemm_gates<<<ggrid, 256, 0, stream>>>(X16, wih16f, b_ih_f, b_hh_f, Gfwd);
    k_gemm_gates<<<ggrid, 256, 0, stream>>>(X16, wih16r, b_ih_r, b_hh_r, Grev);

    k_lstm<<<24, 256, 0, stream>>>(Gfwd, Grev, whh16f, whh16r, hsF, hsR);

    k_head<<<BATCH, 256, 0, stream>>>(hsF, hsR, logits, replace, fcw, fcb, (float*)d_out);
}
